// EnhancedMultiHeadAttention_3350074491572
// MI455X (gfx1250) — matrix operand order repaired and checked by reading, not compiled, not run on hardware
//
#include <hip/hip_runtime.h>

// MHA forward for N=4096, E=512, H=8, D=64 on gfx1250 (wave32, WMMA).
// Pipeline: f32->bf16 convert -> QKV projections (bf16 WMMA, f32 accum)
//           -> fused scores/softmax/attn-write/ctx per 16-row stripe (256KB LDS)
//           -> output projection.
// Outputs (flat, f32): out[N*E], attn[H*N*N], Q[N*E], K[N*E], V[N*E].

typedef __attribute__((ext_vector_type(16))) __bf16 v16bf;
typedef __attribute__((ext_vector_type(8)))  float  v8f;
typedef __attribute__((ext_vector_type(4)))  __bf16 v4bf;

constexpr int N_ = 4096;
constexpr int E_ = 512;
constexpr int H_ = 8;
constexpr int D_ = 64;
constexpr int SROW = 4100;   // padded LDS row stride (floats) to dodge bank conflicts

struct U8x32 { uint4 lo, hi; };

static __device__ __forceinline__ v16bf ld16(const __bf16* p0, const __bf16* p1) {
  U8x32 t;
  t.lo = *reinterpret_cast<const uint4*>(p0);
  t.hi = *reinterpret_cast<const uint4*>(p1);
  return __builtin_bit_cast(v16bf, t);
}

static __device__ __forceinline__ v8f wmma_bf16(v16bf a, v16bf b, v8f c) {
  return __builtin_amdgcn_wmma_f32_16x16x32_bf16(false, a, false, b, (short)0, c,
                                                 false, false);
}

// ---------------------------------------------------------------- convert ----
__global__ void cvt_f32_bf16(const float* __restrict__ src,
                             __bf16* __restrict__ dst, int n4) {
  int i = blockIdx.x * blockDim.x + threadIdx.x;
  if (i >= n4) return;
  float4 v = reinterpret_cast<const float4*>(src)[i];
  v4bf o = { (__bf16)v.x, (__bf16)v.y, (__bf16)v.z, (__bf16)v.w };
  reinterpret_cast<v4bf*>(dst)[i] = o;
}

// ------------------------------------------------- Y = A @ W^T + b (bf16) ----
// A: [M,Kd] bf16 row-major. W: [Nc,Kd] bf16 row-major (so B column n = W row n,
// contiguous along K -> direct B-matrix b128 loads).
// Optionally also writes bf16 Y and bf16 Y^T ([Nc,M], i.e. [H,D,N] for V).
__global__ __launch_bounds__(256) void gemm_bias(
    const __bf16* __restrict__ A, const __bf16* __restrict__ W,
    const float* __restrict__ bias, float* __restrict__ Yf,
    __bf16* __restrict__ Yb, __bf16* __restrict__ Yt,
    int M, int Kd, int Nc) {
  const int lane = threadIdx.x;
  const int lm = lane & 15, lh = lane >> 4;
  const int tm = blockIdx.x;
  const int tn = blockIdx.y * 8 + threadIdx.y;

  const __bf16* pa = A + (size_t)(tm * 16 + lm) * Kd + lh * 8;   // A row, K grp
  const __bf16* pb = W + (size_t)(tn * 16 + lm) * Kd + lh * 8;   // B col, K grp

  v8f c = {0.f, 0.f, 0.f, 0.f, 0.f, 0.f, 0.f, 0.f};
  for (int k0 = 0; k0 < Kd; k0 += 32) {
    v16bf a = ld16(pa + k0, pa + k0 + 16);   // K: lh*8..+7 and 16+lh*8..+7
    v16bf b = ld16(pb + k0, pb + k0 + 16);   // K: lh*8..+7 and 16+lh*8..+7
    c = wmma_bf16(a, b, c);
  }

  const int col = tn * 16 + lm;
  const float bv = bias[col];
  const int row0 = tm * 16 + lh * 8;
#pragma unroll
  for (int i = 0; i < 8; ++i) {
    float v = c[i] + bv;
    Yf[(size_t)(row0 + i) * Nc + col] = v;
    if (Yb) Yb[(size_t)(row0 + i) * Nc + col] = (__bf16)v;
    if (Yt) Yt[(size_t)col * M + (row0 + i)] = (__bf16)v;
  }
}

// ------------------------- fused scores -> softmax -> attn-out -> ctx --------
// One workgroup (8 waves) per (16 query rows, head). Score stripe 16x4096 f32
// lives entirely in LDS (320KB/WGP). Vt is [H,D,N] bf16 so ctx B-loads are
// contiguous along K(=m).
__global__ __launch_bounds__(256) void attn_fused(
    const __bf16* __restrict__ Qb, const __bf16* __restrict__ Kb,
    const __bf16* __restrict__ Vt, float* __restrict__ attnO,
    __bf16* __restrict__ ctxb) {
  extern __shared__ float smem[];
  float* s      = smem;               // 16 * SROW
  float* red    = smem + 16 * SROW;   // 256
  float* rowmax = red + 256;          // 16
  float* invsum = rowmax + 16;        // 16
  float* part   = invsum + 16;        // 4*16*16

  const int tid = threadIdx.x;
  const int w = tid >> 5, lane = tid & 31;
  const int lm = lane & 15, lh = lane >> 4;
  const int br = blockIdx.x, h = blockIdx.y;

  // ---- phase 1: S = (Q Kt) / sqrt(D) into LDS ----
  {
    const __bf16* qp = Qb + (size_t)(br * 16 + lm) * E_ + h * D_;
    v16bf a0 = ld16(qp + lh * 8,      qp + 16 + lh * 8);   // d 0..31
    v16bf a1 = ld16(qp + 32 + lh * 8, qp + 48 + lh * 8);   // d 32..63
    for (int ct = w; ct < N_ / 16; ct += 8) {
      const __bf16* kp = Kb + (size_t)(ct * 16 + lm) * E_ + h * D_ + lh * 8;
      if (ct + 8 < N_ / 16) __builtin_prefetch(kp + (size_t)128 * E_, 0, 0);
      v16bf b0 = ld16(kp, kp + 16);           // d 0..31 slice
      v16bf b1 = ld16(kp + 32, kp + 48);      // d 32..63 slice
      v8f c = {0.f, 0.f, 0.f, 0.f, 0.f, 0.f, 0.f, 0.f};
      c = wmma_bf16(a0, b0, c);
      c = wmma_bf16(a1, b1, c);
      float* sp = s + (size_t)(lh * 8) * SROW + ct * 16 + lm;
#pragma unroll
      for (int i = 0; i < 8; ++i) sp[(size_t)i * SROW] = c[i] * 0.125f;
    }
  }
  __syncthreads();

  // ---- phase 2: softmax over full 4096-row; write normalized attn to HBM ----
  const int r = tid >> 4, cs = tid & 15;
  {
    float mx = -3.402823466e38f;
    for (int j = 0; j < N_ / 16; ++j)
      mx = fmaxf(mx, s[r * SROW + cs + j * 16]);
    red[r * 16 + cs] = mx;
    __syncthreads();
    if (tid < 16) {
      float m = red[tid * 16];
#pragma unroll
      for (int j = 1; j < 16; ++j) m = fmaxf(m, red[tid * 16 + j]);
      rowmax[tid] = m;
    }
    __syncthreads();
    const float rm = rowmax[r];
    float sum = 0.f;
    for (int j = 0; j < N_ / 16; ++j) {
      int idx = r * SROW + cs + j * 16;
      float e = __expf(s[idx] - rm);
      s[idx] = e;                 // keep un-normalized exp in LDS for ctx GEMM
      sum += e;
    }
    red[r * 16 + cs] = sum;
    __syncthreads();
    if (tid < 16) {
      float t = 0.f;
#pragma unroll
      for (int j = 0; j < 16; ++j) t += red[tid * 16 + j];
      invsum[tid] = 1.f / t;
    }
    __syncthreads();
    const float inv = invsum[r];
    float* ar = attnO + ((size_t)h * N_ + br * 16 + r) * N_;
    for (int j = 0; j < N_ / 16; ++j) {
      int col = cs + j * 16;
      ar[col] = s[r * SROW + col] * inv;   // normalize on the fly
    }
  }

  // ---- phase 3: ctx = attn @ V (A from LDS, B from Vt), fold 1/sum at end ----
  {
    const int dt = w & 3;     // which 16-wide d tile
    const int mh = w >> 2;    // which half of the m range
    const __bf16* vp =
        Vt + (size_t)(h * D_ + dt * 16 + lm) * N_ + mh * 2048 + lh * 8;
    const float* sr = s + (size_t)lm * SROW + mh * 2048;

    v8f c = {0.f, 0.f, 0.f, 0.f, 0.f, 0.f, 0.f, 0.f};
    for (int it = 0; it < 64; ++it) {
      const int o0 = it * 32 + lh * 8;
      float4 q0 = *reinterpret_cast<const float4*>(sr + o0);
      float4 q1 = *reinterpret_cast<const float4*>(sr + o0 + 4);
      float4 q2 = *reinterpret_cast<const float4*>(sr + o0 + 16);
      float4 q3 = *reinterpret_cast<const float4*>(sr + o0 + 20);
      v16bf a = {(__bf16)q0.x, (__bf16)q0.y, (__bf16)q0.z, (__bf16)q0.w,
                 (__bf16)q1.x, (__bf16)q1.y, (__bf16)q1.z, (__bf16)q1.w,
                 (__bf16)q2.x, (__bf16)q2.y, (__bf16)q2.z, (__bf16)q2.w,
                 (__bf16)q3.x, (__bf16)q3.y, (__bf16)q3.z, (__bf16)q3.w};
      v16bf b = ld16(vp + it * 32, vp + it * 32 + 16);
      c = wmma_bf16(a, b, c);
    }
    if (mh == 1) {
#pragma unroll
      for (int i = 0; i < 8; ++i)
        part[dt * 256 + (i + lh * 8) * 16 + lm] = c[i];
    }
    __syncthreads();
    if (mh == 0) {
#pragma unroll
      for (int i = 0; i < 8; ++i) {
        const int Ml = i + lh * 8;
        float v = (c[i] + part[dt * 256 + Ml * 16 + lm]) * invsum[Ml];
        ctxb[(size_t)(br * 16 + Ml) * E_ + h * D_ + dt * 16 + lm] = (__bf16)v;
      }
    }
  }
}

// ------------------------------------------------------------------ launch ---
extern "C" void kernel_launch(void* const* d_in, const int* in_sizes, int n_in,
                              void* d_out, int out_size, void* d_ws,
                              size_t ws_size, hipStream_t stream) {
  (void)in_sizes; (void)n_in; (void)out_size; (void)ws_size;

  const float* x  = (const float*)d_in[0];
  const float* Wq = (const float*)d_in[1];
  const float* bq = (const float*)d_in[2];
  const float* Wk = (const float*)d_in[3];
  const float* bk = (const float*)d_in[4];
  const float* Wv = (const float*)d_in[5];
  const float* bv = (const float*)d_in[6];
  const float* Wo = (const float*)d_in[7];
  const float* bo = (const float*)d_in[8];

  float* out  = (float*)d_out;
  float* attn = out + (size_t)N_ * E_;
  float* Qf   = attn + (size_t)H_ * N_ * N_;
  float* Kf   = Qf + (size_t)N_ * E_;
  float* Vf   = Kf + (size_t)N_ * E_;

  char* wsp = (char*)d_ws;
  __bf16* xb   = (__bf16*)wsp; wsp += (size_t)N_ * E_ * 2;
  __bf16* Wqb  = (__bf16*)wsp; wsp += (size_t)E_ * E_ * 2;
  __bf16* Wkb  = (__bf16*)wsp; wsp += (size_t)E_ * E_ * 2;
  __bf16* Wvb  = (__bf16*)wsp; wsp += (size_t)E_ * E_ * 2;
  __bf16* Wob  = (__bf16*)wsp; wsp += (size_t)E_ * E_ * 2;
  __bf16* Qb   = (__bf16*)wsp; wsp += (size_t)N_ * E_ * 2;
  __bf16* Kbb  = (__bf16*)wsp; wsp += (size_t)N_ * E_ * 2;
  __bf16* Vtb  = (__bf16*)wsp; wsp += (size_t)N_ * E_ * 2;  // [H,D,N]
  __bf16* ctxb = (__bf16*)wsp; wsp += (size_t)N_ * E_ * 2;  // [N,E]

  // f32 -> bf16
  cvt_f32_bf16<<<(N_ * E_ / 4 + 255) / 256, 256, 0, stream>>>(x, xb, N_ * E_ / 4);
  cvt_f32_bf16<<<(E_ * E_ / 4 + 255) / 256, 256, 0, stream>>>(Wq, Wqb, E_ * E_ / 4);
  cvt_f32_bf16<<<(E_ * E_ / 4 + 255) / 256, 256, 0, stream>>>(Wk, Wkb, E_ * E_ / 4);
  cvt_f32_bf16<<<(E_ * E_ / 4 + 255) / 256, 256, 0, stream>>>(Wv, Wvb, E_ * E_ / 4);
  cvt_f32_bf16<<<(E_ * E_ / 4 + 255) / 256, 256, 0, stream>>>(Wo, Wob, E_ * E_ / 4);

  // Q/K/V projections (f32 outputs to d_out, bf16 side copies for attention)
  dim3 gproj(N_ / 16, (E_ / 16) / 8);
  dim3 bproj(32, 8);
  gemm_bias<<<gproj, bproj, 0, stream>>>(xb, Wqb, bq, Qf, Qb, nullptr, N_, E_, E_);
  gemm_bias<<<gproj, bproj, 0, stream>>>(xb, Wkb, bk, Kf, Kbb, nullptr, N_, E_, E_);
  gemm_bias<<<gproj, bproj, 0, stream>>>(xb, Wvb, bv, Vf, nullptr, Vtb, N_, E_, E_);

  // fused attention (16 rows x full 4096 cols per workgroup, 256KB+ LDS)
  const size_t shmem =
      (size_t)(16 * SROW + 256 + 16 + 16 + 4 * 16 * 16) * sizeof(float);
  hipFuncSetAttribute(reinterpret_cast<const void*>(attn_fused),
                      hipFuncAttributeMaxDynamicSharedMemorySize, (int)shmem);
  attn_fused<<<dim3(N_ / 16, H_), 256, shmem, stream>>>(Qb, Kbb, Vtb, attn, ctxb);

  // output projection
  gemm_bias<<<gproj, bproj, 0, stream>>>(ctxb, Wob, bo, out, nullptr, nullptr,
                                         N_, E_, E_);
}
